// SelfAttention_37855841747255
// MI455X (gfx1250) — matrix operand order repaired and checked by reading, not compiled, not run on hardware
//
#include <hip/hip_runtime.h>
#include <hip/hip_bf16.h>

#define NUM_HEADS 16
#define EMBED_DIM 1024
#define HEAD_SIZE 64
#define BATCH 4
#define SEQ 2048

typedef __bf16 v4bf  __attribute__((ext_vector_type(4)));
typedef __bf16 v8bf  __attribute__((ext_vector_type(8)));
typedef __bf16 v16bf __attribute__((ext_vector_type(16)));
typedef float  v8f   __attribute__((ext_vector_type(8)));
typedef unsigned int u32x4 __attribute__((ext_vector_type(4)));
typedef int          i32x4 __attribute__((ext_vector_type(4)));
typedef int          i32x8 __attribute__((ext_vector_type(8)));

static __device__ __forceinline__ v16bf cat16(v8bf lo, v8bf hi) {
  return __builtin_shufflevector(lo, hi, 0,1,2,3,4,5,6,7,8,9,10,11,12,13,14,15);
}

static __device__ __forceinline__ v8f wmma_bf16(v16bf a, v16bf b, v8f c) {
  // D = A(16x32 bf16) * B(32x16 bf16) + C(16x16 f32)
  return __builtin_amdgcn_wmma_f32_16x16x32_bf16(false, a, false, b, (short)0, c,
                                                 false, false);
}

static __device__ __forceinline__ float rmax16(float v) {
  v = fmaxf(v, __shfl_xor(v, 1, 32));
  v = fmaxf(v, __shfl_xor(v, 2, 32));
  v = fmaxf(v, __shfl_xor(v, 4, 32));
  v = fmaxf(v, __shfl_xor(v, 8, 32));
  return v;
}
static __device__ __forceinline__ float rsum16(float v) {
  v += __shfl_xor(v, 1, 32);
  v += __shfl_xor(v, 2, 32);
  v += __shfl_xor(v, 4, 32);
  v += __shfl_xor(v, 8, 32);
  return v;
}

// ---------------------------------------------------------------- converters
__global__ void cvt_f32_bf16(const float* __restrict__ in,
                             __bf16* __restrict__ out, int n4) {
  int i = blockIdx.x * blockDim.x + threadIdx.x;
  if (i >= n4) return;
  float4 v = ((const float4*)in)[i];
  v4bf o;
  o[0] = (__bf16)v.x; o[1] = (__bf16)v.y; o[2] = (__bf16)v.z; o[3] = (__bf16)v.w;
  ((v4bf*)out)[i] = o;
}

// WT[n][k] = (bf16) W[k][n]   (D x D), LDS-tiled transpose
__global__ void transpose_cvt(const float* __restrict__ W,
                              __bf16* __restrict__ WT) {
  __shared__ float t[32][33];
  const int n0 = blockIdx.x * 32, k0 = blockIdx.y * 32;
  const int tx = threadIdx.x, ty = threadIdx.y;   // (32, 8)
#pragma unroll
  for (int i = 0; i < 4; ++i)
    t[ty + 8 * i][tx] = W[(size_t)(k0 + ty + 8 * i) * EMBED_DIM + n0 + tx];
  __syncthreads();
#pragma unroll
  for (int i = 0; i < 4; ++i)
    WT[(size_t)(n0 + ty + 8 * i) * EMBED_DIM + k0 + tx] =
        (__bf16)t[tx][ty + 8 * i];
}

// ------------------------------------------------ TDM B-tile staging helpers
#define HAVE_TDM __has_builtin(__builtin_amdgcn_tensor_load_to_lds)

// Stage BT[n0..n0+63][k0..k0+31] (bf16, row stride K elements) into LDS buf
// [64][32] row-major. TDM path: one descriptor issued by wave 0 only.
static __device__ __forceinline__ void stage_b_issue(
    __bf16* __restrict__ lds_dst, const __bf16* __restrict__ BT,
    int n0, int k0, int K, int tid) {
#if HAVE_TDM
  if (tid < 32) {
    const unsigned lds = (unsigned)(uintptr_t)lds_dst;
    const unsigned long long ga =
        (unsigned long long)(uintptr_t)(BT + (size_t)n0 * K + k0);
    u32x4 g0;
    g0[0] = 1u;                                   // count=1, user descriptor
    g0[1] = lds;                                  // lds_addr [63:32]
    g0[2] = (unsigned)ga;                         // global_addr [95:64]
    g0[3] = (unsigned)((ga >> 32) & 0x1FFFFFFu)   // global_addr [120:96]
            | (2u << 30);                         // type=2 (image)
    i32x8 g1;
    g1[0] = (int)(1u << 16);        // wg_mask=0, data_size=1 (2B)
    g1[1] = (int)(32u << 16);       // tensor_dim0[15:0]=32 in bits 63:48
    g1[2] = (int)(64u << 16);       // tensor_dim0 hi=0 | tensor_dim1=64
    g1[3] = (int)(32u << 16);       // tensor_dim1 hi=0 | tile_dim0=32
    g1[4] = 64;                     // tile_dim1=64, tile_dim2=0
    g1[5] = K;                      // tensor_dim0_stride lo (elements)
    g1[6] = 0;                      // stride0 hi | stride1 lo
    g1[7] = 0;                      // stride1 hi
    const i32x4 z4 = {0, 0, 0, 0};
#if defined(__clang_major__) && (__clang_major__ >= 23)
    const i32x8 z8 = {0, 0, 0, 0, 0, 0, 0, 0};
    __builtin_amdgcn_tensor_load_to_lds(g0, g1, z4, z4, z8, 0);
#else
    __builtin_amdgcn_tensor_load_to_lds(g0, g1, z4, z4, 0);
#endif
  }
#else
  // Fallback: cooperative global->LDS copy by all 128 threads (2 x b128 each)
#pragma unroll
  for (int p = 0; p < 2; ++p) {
    const int c = p * 128 + tid;        // 8-half chunk id, 256 chunks total
    const int r = c >> 2;               // row 0..63
    const int c8 = (c & 3) * 8;         // half offset in row
    *(v8bf*)(lds_dst + r * 32 + c8) =
        *(const v8bf*)(BT + (size_t)(n0 + r) * K + k0 + c8);
  }
#endif
}

static __device__ __forceinline__ void stage_b_fence(int tid) {
#if HAVE_TDM
  if (tid < 32) __builtin_amdgcn_s_wait_tensorcnt(0);
#endif
  __syncthreads();
}

// -------------------------------------------------------------------- GEMM
// out = (A[MxK]bf16 @ BT[NxK]bf16^T + bias[N]) * scale
// B tile (shared by the block's 4 waves) is staged in LDS via TDM, double-
// buffered across the K loop. A fragments stream from global (L2 resident).
// mode 0: bf16 out[b][h][s][d]      (Q / K, scale folds 1/sqrt(Hd) into Q)
// mode 1: bf16 out[b][h][d][s]      (V transposed)
// mode 2: f32  out[m][n]            (final projection)
__global__ __launch_bounds__(128)
void gemm_bf16(const __bf16* __restrict__ A, const __bf16* __restrict__ BT,
               const float* __restrict__ bias, void* __restrict__ outp,
               int M, int N, int K, float scale, int mode) {
  __shared__ __bf16 bstage[2][64 * 32];
  const int tid  = threadIdx.x;
  const int lane = tid & 31;
  const int wave = tid >> 5;
  const int m0 = blockIdx.x * 256 + wave * 64;
  const int n0 = blockIdx.y * 64;
  const int row  = lane & 15;
  const int half = lane >> 4;
  const int kb8  = half * 8;
  const int kb16 = half * 16;

  v8f acc[4][4] = {};

  // prime buffer 0
  stage_b_issue(&bstage[0][0], BT, n0, 0, K, tid);
  stage_b_fence(tid);

  for (int k0 = 0; k0 < K; k0 += 32) {
    const int cur = (k0 >> 5) & 1;
    if (k0 + 32 < K)
      stage_b_issue(&bstage[cur ^ 1][0], BT, n0, k0 + 32, K, tid);

    v16bf af[4], bfr[4];
#pragma unroll
    for (int i = 0; i < 4; ++i) {
      const __bf16* p = A + (size_t)(m0 + 16 * i + row) * K + k0;
      af[i] = cat16(*(const v8bf*)(p + kb16), *(const v8bf*)(p + kb16 + 8));
    }
#pragma unroll
    for (int j = 0; j < 4; ++j) {
      const __bf16* p = &bstage[cur][(16 * j + row) * 32 + kb16];
      bfr[j] = cat16(*(const v8bf*)p, *(const v8bf*)(p + 8));
    }
#pragma unroll
    for (int i = 0; i < 4; ++i)
#pragma unroll
      for (int j = 0; j < 4; ++j)
        acc[i][j] = wmma_bf16(af[i], bfr[j], acc[i][j]);

    stage_b_fence(tid);   // next buffer ready; all waves done with cur
  }

#pragma unroll
  for (int j = 0; j < 4; ++j) {
    const int gn = n0 + 16 * j + row;       // C/D layout: N = lane&15 (+tile)
    const float bv = bias[gn];
    const int h = gn >> 6, d = gn & 63;
#pragma unroll
    for (int i = 0; i < 4; ++i) {
#pragma unroll
      for (int jv = 0; jv < 8; ++jv) {
        const int gm = m0 + 16 * i + half * 8 + jv;   // M = vgpr + 8*half
        const float val = (acc[i][j][jv] + bv) * scale;
        if (mode == 2) {
          ((float*)outp)[(size_t)gm * N + gn] = val;
        } else {
          const int b = gm >> 11;            // / SEQ
          const int s = gm & (SEQ - 1);
          const size_t idx = (mode == 0)
              ? (((size_t)(b * NUM_HEADS + h) * SEQ + s) * HEAD_SIZE + d)
              : (((size_t)(b * NUM_HEADS + h) * HEAD_SIZE + d) * SEQ + s);
          ((__bf16*)outp)[idx] = (__bf16)val;
        }
      }
    }
  }
}

// --------------------------------------------------------- flash attention
// One wave handles 16 query rows of one (b,h). Q already scaled by 1/sqrt(Hd).
// Q,K: [b*h][s][64] bf16.  VT: [b*h][64][s] bf16.  ctx: [b][s][1024] bf16.
__global__ __launch_bounds__(32)
void attn_fwd(const __bf16* __restrict__ Q, const __bf16* __restrict__ Km,
              const __bf16* __restrict__ VT, __bf16* __restrict__ ctx) {
  __shared__ __bf16 pl[16 * 32];            // probs staging (row-major 16x32)
  const int lane = threadIdx.x & 31;
  const int row  = lane & 15;
  const int half = lane >> 4;
  const int kb8  = half * 8;
  const int kb16 = half * 16;

  const int qt = blockIdx.x & (SEQ / 16 - 1);   // 128 q-tiles
  const int bh = blockIdx.x >> 7;
  const int q0 = qt * 16;

  // Q A-fragments (16x32 each, covering Hd = 64)
  const __bf16* qp = Q + ((size_t)bh * SEQ + q0 + row) * HEAD_SIZE;
  const v16bf qf0 = cat16(*(const v8bf*)(qp + kb16),      *(const v8bf*)(qp + kb16 + 8));
  const v16bf qf1 = cat16(*(const v8bf*)(qp + 32 + kb16), *(const v8bf*)(qp + 40 + kb16));

  float mrow[8], lrow[8];
#pragma unroll
  for (int j = 0; j < 8; ++j) { mrow[j] = -3.0e38f; lrow[j] = 0.f; }
  v8f acc[4] = {};

  for (int s0 = 0; s0 < SEQ; s0 += 32) {
    // ---- scores for 32 keys (two 16-key tiles), K-dim = Hd = 64 ----
    v8f sc[2];
#pragma unroll
    for (int t = 0; t < 2; ++t) {
      const __bf16* kp =
          Km + ((size_t)bh * SEQ + s0 + 16 * t + row) * HEAD_SIZE + kb16;
      const v16bf kf0 = cat16(*(const v8bf*)kp,        *(const v8bf*)(kp + 8));
      const v16bf kf1 = cat16(*(const v8bf*)(kp + 32), *(const v8bf*)(kp + 40));
      v8f c = {};
      c = wmma_bf16(qf0, kf0, c);
      c = wmma_bf16(qf1, kf1, c);
      sc[t] = c;
    }
    // ---- online softmax over this 32-key panel ----
#pragma unroll
    for (int j = 0; j < 8; ++j) {
      const float tm = rmax16(fmaxf(sc[0][j], sc[1][j]));
      const float mn = fmaxf(mrow[j], tm);
      const float corr = __expf(mrow[j] - mn);
      mrow[j] = mn;
      const float p0 = __expf(sc[0][j] - mn);
      const float p1 = __expf(sc[1][j] - mn);
      lrow[j] = lrow[j] * corr + rsum16(p0 + p1);
#pragma unroll
      for (int nt = 0; nt < 4; ++nt) acc[nt][j] *= corr;
      const int jr = half * 8 + j;          // C layout row
      pl[jr * 32 + row]      = (__bf16)p0;  // key col = lane&15 (+16 for t=1)
      pl[jr * 32 + 16 + row] = (__bf16)p1;
    }
    asm volatile("s_wait_dscnt 0" ::: "memory");   // LDS RAW before A-frag read
    // probs as A-fragment (16 rows x 32 keys)
    const v16bf pf = cat16(*(const v8bf*)&pl[row * 32 + kb16],
                           *(const v8bf*)&pl[row * 32 + kb16 + 8]);
    // ---- ctx += P @ V  (N = head dim, Bt = VT rows are contiguous) ----
#pragma unroll
    for (int nt = 0; nt < 4; ++nt) {
      const __bf16* vp =
          VT + ((size_t)bh * HEAD_SIZE + nt * 16 + row) * SEQ + s0 + kb16;
      const v16bf vf = cat16(*(const v8bf*)vp, *(const v8bf*)(vp + 8));
      acc[nt] = wmma_bf16(pf, vf, acc[nt]);
    }
  }

  const int b = bh / NUM_HEADS, h = bh - b * NUM_HEADS;
#pragma unroll
  for (int nt = 0; nt < 4; ++nt) {
#pragma unroll
    for (int j = 0; j < 8; ++j) {
      const int r = half * 8 + j;
      const float val = acc[nt][j] / lrow[j];
      ctx[((size_t)(b * SEQ + q0 + r)) * EMBED_DIM + h * HEAD_SIZE + nt * 16 + row] =
          (__bf16)val;
    }
  }
}

// ------------------------------------------------------------------- launch
extern "C" void kernel_launch(void* const* d_in, const int* in_sizes, int n_in,
                              void* d_out, int out_size, void* d_ws, size_t ws_size,
                              hipStream_t stream) {
  const float* x  = (const float*)d_in[0];
  const float* Wq = (const float*)d_in[1];
  const float* bq = (const float*)d_in[2];
  const float* Wk = (const float*)d_in[3];
  const float* bk = (const float*)d_in[4];
  const float* Wv = (const float*)d_in[5];
  const float* bv = (const float*)d_in[6];
  const float* Wo = (const float*)d_in[7];
  const float* bo = (const float*)d_in[8];

  const int M = BATCH * SEQ;      // 8192
  const int N = EMBED_DIM;        // 1024
  const int K = EMBED_DIM;        // 1024

  char* ws = (char*)d_ws;
  size_t off = 0;
  const size_t XB = (size_t)M * EMBED_DIM * sizeof(__bf16);   // 16 MB
  const size_t WB = (size_t)EMBED_DIM * EMBED_DIM * sizeof(__bf16);
  __bf16* xb  = (__bf16*)(ws + off); off += XB;
  __bf16* wqT = (__bf16*)(ws + off); off += WB;
  __bf16* wkT = (__bf16*)(ws + off); off += WB;
  __bf16* wvT = (__bf16*)(ws + off); off += WB;
  __bf16* woT = (__bf16*)(ws + off); off += WB;
  __bf16* qb  = (__bf16*)(ws + off); off += XB;
  __bf16* kb  = (__bf16*)(ws + off); off += XB;
  __bf16* vT  = (__bf16*)(ws + off); off += XB;
  __bf16* ctx = (__bf16*)(ws + off); off += XB;

  // bf16 conversions
  const int n4 = M * EMBED_DIM / 4;
  cvt_f32_bf16<<<(n4 + 255) / 256, 256, 0, stream>>>(x, xb, n4);
  dim3 tgrid(EMBED_DIM / 32, EMBED_DIM / 32), tblk(32, 8);
  transpose_cvt<<<tgrid, tblk, 0, stream>>>(Wq, wqT);
  transpose_cvt<<<tgrid, tblk, 0, stream>>>(Wk, wkT);
  transpose_cvt<<<tgrid, tblk, 0, stream>>>(Wv, wvT);
  transpose_cvt<<<tgrid, tblk, 0, stream>>>(Wo, woT);

  // QKV projections (scale 1/sqrt(Hd) folded into Q)
  dim3 ggrid(M / 256, N / 64), gblk(128);
  gemm_bf16<<<ggrid, gblk, 0, stream>>>(xb, wqT, bq, qb, M, N, K, 0.125f, 0);
  gemm_bf16<<<ggrid, gblk, 0, stream>>>(xb, wkT, bk, kb, M, N, K, 1.0f,   0);
  gemm_bf16<<<ggrid, gblk, 0, stream>>>(xb, wvT, bv, vT, M, N, K, 1.0f,   1);

  // flash attention: one wave per (b,h,16-row q tile)
  attn_fwd<<<BATCH * NUM_HEADS * (SEQ / 16), 32, 0, stream>>>(qb, kb, vT, ctx);

  // output projection -> f32
  gemm_bf16<<<ggrid, gblk, 0, stream>>>(ctx, woT, bo, (float*)d_out,
                                        M, N, K, 1.0f, 2);
}
